// HawkesDecoder_74345883894331
// MI455X (gfx1250) — matrix operand order repaired and checked by reading, not compiled, not run on hardware
//
#include <hip/hip_runtime.h>
#include <math.h>

#define DEVFN __device__ __forceinline__

typedef __bf16 bf16_t;
typedef __bf16 bfx16 __attribute__((ext_vector_type(16)));
typedef __bf16 bfx8  __attribute__((ext_vector_type(8)));
typedef float  fx8   __attribute__((ext_vector_type(8)));

// ---- problem constants -----------------------------------------------------
constexpr int B = 32, T = 512, H = 512, E = 256, L = 64;
constexpr int RIN  = E + L;        // 320 precomputable feed columns
constexpr int IN   = RIN + H;      // 832
constexpr int G7   = 7 * H;        // 3584 gate columns
constexpr int OUT5 = 5 * H;        // 2560 output columns per (b,t)
constexpr int KT     = IN / 32;    // 26 k-tiles of 32
constexpr int KT_IN  = RIN / 32;   // 10 (input part, streamed from L2)
constexpr int KT_REC = H / 32;     // 16 (recurrent part, LDS resident)
constexpr int NWG = 32;            // persistent workgroups, each owns 16 j-cols
constexpr int NJ  = 16;
constexpr int BLK = 448;           // 14 waves = (mt 0..1) x (gate 0..6)

// ---- workspace layout (bytes) ---------------------------------------------
constexpr size_t OFF_BAR = 0;                                   // barrier state
constexpr size_t OFF_DT  = 256;                                 // [B,T] f32
constexpr size_t OFF_HD  = OFF_DT + (size_t)B * T * 4;          // [2,B,H] bf16 ping-pong
constexpr size_t OFF_CD  = OFF_HD + (size_t)2 * B * H * 2;      // [B,H] f32
constexpr size_t OFF_CB  = OFF_CD + (size_t)B * H * 4;          // [B,H] f32
constexpr size_t OFF_RF  = OFF_CB + (size_t)B * H * 4;          // A-fragments of R
constexpr size_t OFF_WF  = OFF_RF + (size_t)T * 2 * KT_IN * 512 * 2; // B-fragments of W

// ---- math helpers ----------------------------------------------------------
DEVFN float sigmoid_(float x) { return 1.f / (1.f + __expf(-x)); }
DEVFN float tanh_(float x) {
  float xx = fminf(fmaxf(x, -15.f), 15.f);
  float e = __expf(2.f * xx);
  return (e - 1.f) / (e + 1.f);
}
DEVFN float softplus_(float x) { return x > 20.f ? x : log1pf(__expf(x)); }

DEVFN fx8 wmma_bf16(bfx16 a, bfx16 b, fx8 c) {
  return __builtin_amdgcn_wmma_f32_16x16x32_bf16(false, a, false, b, (short)0, c, false, false);
}

// Input-K partial GEMM (independent of the recurrence; runs one step ahead).
DEVFN void inputGemm(const bf16_t* __restrict__ rbase, const bf16_t* __restrict__ wbase,
                     fx8& ip0, fx8& ip1, fx8& ip2, fx8& ip3) {
  ip0 = fx8{}; ip1 = fx8{}; ip2 = fx8{}; ip3 = fx8{};
  #pragma unroll
  for (int kt = 0; kt < KT_IN; ++kt) {
    bfx16 a = *(const bfx16*)(rbase + ((size_t)kt << 9));
    bfx16 b = *(const bfx16*)(wbase + ((size_t)kt << 9));
    switch (kt & 3) {
      case 0: ip0 = wmma_bf16(a, b, ip0); break;
      case 1: ip1 = wmma_bf16(a, b, ip1); break;
      case 2: ip2 = wmma_bf16(a, b, ip2); break;
      default: ip3 = wmma_bf16(a, b, ip3); break;
    }
  }
}

// ---- grid rendezvous (32 persistent WGs) -----------------------------------
DEVFN void gridBarrier(unsigned* bar, unsigned nwg) {
  __threadfence();
  __syncthreads();
  if (threadIdx.x == 0) {
    volatile unsigned* vcnt = bar;
    volatile unsigned* vgen = bar + 1;
    unsigned g = *vgen;
    unsigned old = atomicAdd(bar, 1u);
    if (old == nwg - 1) {
      *vcnt = 0;
      __threadfence();
      atomicAdd(bar + 1, 1u);
    } else {
      while (*vgen == g) __builtin_amdgcn_s_sleep(2);
    }
    __threadfence();
  }
  __syncthreads();
}

// ---- prologue kernels ------------------------------------------------------
__global__ void k_zero(unsigned* bar) { if (threadIdx.x < 64) bar[threadIdx.x] = 0u; }

__global__ void k_dt(const float* __restrict__ ts, float* __restrict__ dts) {
  int i = blockIdx.x * 256 + threadIdx.x;
  if (i >= B * T) return;
  int b = i / T, t = i % T;
  dts[i] = (t == 0) ? ts[b * T] : ts[b * T + t] - ts[b * T + t - 1];
}

// Pack R = [emb(marks) | latent] into WMMA A-fragment layout (bf16 16x32):
// lane m = lane&15, hi = lane>>4; element e -> K = kt*32 + (e>>3)*16 + hi*8 + (e&7)
__global__ __launch_bounds__(512) void k_pack_r(const int* __restrict__ marks,
                                                const float* __restrict__ latent,
                                                const float* __restrict__ emb,
                                                bf16_t* __restrict__ rfrag) {
  int bid = blockIdx.x;                 // = (t*2 + mt)*KT_IN + kt
  int t = bid / (2 * KT_IN), r = bid % (2 * KT_IN), mt = r / KT_IN, kt = r % KT_IN;
  int lane = threadIdx.x >> 4, e = threadIdx.x & 15;
  int m  = mt * 16 + (lane & 15);
  int hi = lane >> 4;
  int K  = kt * 32 + ((e >> 3) << 4) + hi * 8 + (e & 7);
  float v = (K < E) ? emb[(size_t)marks[m * T + t] * E + K] : latent[m * L + (K - E)];
  rfrag[((size_t)bid << 9) + threadIdx.x] = (bf16_t)v;
}

// Pack W_cell into WMMA B-fragment layout (bf16 32x16):
// lane n = lane&15, hi = lane>>4; element e -> K = kt*32 + (e>>3)*16 + hi*8 + (e&7)
__global__ __launch_bounds__(512) void k_pack_w(const float* __restrict__ wcell,
                                                bf16_t* __restrict__ wfrag) {
  int bid = blockIdx.x;                 // = (wg*7 + nt)*KT + kt
  int wg = bid / (7 * KT), r = bid % (7 * KT), nt = r / KT, kt = r % KT;
  int lane = threadIdx.x >> 4, e = threadIdx.x & 15;
  int nl = lane & 15, hi = lane >> 4;
  int K = kt * 32 + ((e >> 3) << 4) + hi * 8 + (e & 7);
  int n = nt * H + wg * NJ + nl;
  wfrag[((size_t)bid << 9) + threadIdx.x] = (bf16_t)wcell[(size_t)K * G7 + n];
}

// init = latent @ W_init + b_init; write state0 row + initial carries
__global__ __launch_bounds__(128) void k_init(const float* __restrict__ latent,
                                              const float* __restrict__ winit,
                                              const float* __restrict__ binit,
                                              float* __restrict__ out,
                                              bf16_t* __restrict__ hdbuf,
                                              float* __restrict__ cd0,
                                              float* __restrict__ cb0) {
  int c = blockIdx.x * 128 + threadIdx.x;   // 0..6H-1
  int b = blockIdx.y;
  float acc = binit[c];
  for (int l = 0; l < L; ++l) acc = fmaf(latent[b * L + l], winit[(size_t)l * (6 * H) + c], acc);
  int chunk = c / H, j = c % H;
  float* o = out + (size_t)b * (T + 1) * OUT5;   // row t=0
  switch (chunk) {
    case 0: { float h = tanh_(acc); o[j] = h; hdbuf[b * H + j] = (bf16_t)h; break; }
    case 1: cd0[b * H + j] = tanh_(acc); break;
    case 2: { float v = tanh_(acc); cb0[b * H + j] = v; o[2 * H + j] = v; break; }
    case 3: o[3 * H + j] = tanh_(acc); break;
    case 4: o[4 * H + j] = softplus_(acc); break;
    case 5: o[H + j] = sigmoid_(acc); break;
  }
}

// ---- persistent recurrent kernel -------------------------------------------
__global__ __launch_bounds__(BLK) void hawkes_recurrent(
    const bf16_t* __restrict__ rfrag, const bf16_t* __restrict__ wfrag,
    const float* __restrict__ dts, const float* __restrict__ bcell,
    bf16_t* __restrict__ hdbuf, const float* __restrict__ cd0,
    const float* __restrict__ cb0, float* __restrict__ out,
    unsigned* __restrict__ bar) {
  __shared__ __attribute__((aligned(32))) bf16_t s_w[7 * KT_REC * 512]; // 114688 B
  __shared__ float s_g[7 * B * NJ];                                     // 14336 B
  __shared__ float s_cd[B * NJ];
  __shared__ float s_cb[B * NJ];

  const int tid  = threadIdx.x;
  const int wg   = blockIdx.x;       // owns j in [wg*16, wg*16+16)
  const int lane = tid & 31;
  const int wave = tid >> 5;         // 0..13
  const int mt   = wave & 1;         // batch-row tile
  const int g    = wave >> 1;        // gate group 0..6
  const int hi   = lane >> 4, nl = lane & 15;

  // Stage recurrent-K weight fragments into LDS (resident all 512 steps).
  for (int i = tid; i < 7 * KT_REC * 64; i += BLK) {
    int frag = i >> 6, q = i & 63;
    int nt = frag >> 4, kr = frag & 15;
    const uint4* s = (const uint4*)(wfrag + (((size_t)(wg * 7 + nt) * KT + (kr + KT_IN)) << 9));
    uint4* d = (uint4*)(s_w + (size_t)(nt * KT_REC + kr) * 512);
    d[q] = s[q];
  }
  // Stage initial carries for this WG's j slice.
  for (int p = tid; p < B * NJ; p += BLK) {
    int m = p >> 4, jl = p & 15;
    s_cd[p] = cd0[m * H + wg * NJ + jl];
    s_cb[p] = cb0[m * H + wg * NJ + jl];
  }
  __syncthreads();

  const float bias = bcell[g * H + wg * NJ + nl];
  const int   m0   = mt * 16 + nl;                       // this lane's A row
  const bf16_t* wbase = wfrag + (((size_t)(wg * 7 + g) * KT) << 9) + lane * 16;
  const bf16_t* sbase = s_w + (size_t)(g * KT_REC) * 512 + lane * 16;  // LDS

  // Pre-compute input-K partial sums for t = 0 (one step ahead of recurrence).
  fx8 ip0, ip1, ip2, ip3;
  inputGemm(rfrag + ((size_t)mt * KT_IN << 9) + lane * 16, wbase, ip0, ip1, ip2, ip3);

  int pb = 0;  // h_d ping-pong index
  for (int t = 0; t < T; ++t) {
    // ---- recurrent-K part: A = h_d (global/L2), B resident in LDS ---------
    fx8 acc0 = ip0, acc1 = ip1, acc2 = ip2, acc3 = ip3;
    const bf16_t* hbase = hdbuf + (size_t)pb * B * H + m0 * H + hi * 8;
    #pragma unroll
    for (int kr = 0; kr < KT_REC; ++kr) {
      union { bfx16 v; bfx8 h[2]; } u;
      u.h[0] = *(const bfx8*)(hbase + kr * 32);
      u.h[1] = *(const bfx8*)(hbase + kr * 32 + 16);
      bfx16 b = *(const bfx16*)(sbase + (size_t)kr * 512);
      switch (kr & 3) {
        case 0: acc0 = wmma_bf16(u.v, b, acc0); break;
        case 1: acc1 = wmma_bf16(u.v, b, acc1); break;
        case 2: acc2 = wmma_bf16(u.v, b, acc2); break;
        default: acc3 = wmma_bf16(u.v, b, acc3); break;
      }
    }
    fx8 acc = (acc0 + acc1) + (acc2 + acc3);

    // D layout: lane n = lane&15, VGPR v -> m = v + 8*(lane>>4) (within 16-row tile)
    #pragma unroll
    for (int v = 0; v < 8; ++v) {
      int m = mt * 16 + v + 8 * hi;
      s_g[(g * B + m) * NJ + nl] = acc[v] + bias;
    }
    __syncthreads();

    bf16_t* hdn = hdbuf + (size_t)(pb ^ 1) * B * H;
    for (int q = tid; q < B * NJ; q += BLK) {
      int m = q >> 4, jl = q & 15;
      float gi  = sigmoid_ (s_g[(0 * B + m) * NJ + jl]);
      float gf  = sigmoid_ (s_g[(1 * B + m) * NJ + jl]);
      float gz  = tanh_    (s_g[(2 * B + m) * NJ + jl]);
      float go  = sigmoid_ (s_g[(3 * B + m) * NJ + jl]);
      float gib = sigmoid_ (s_g[(4 * B + m) * NJ + jl]);
      float gfb = sigmoid_ (s_g[(5 * B + m) * NJ + jl]);
      float gd  = softplus_(s_g[(6 * B + m) * NJ + jl]);
      float dt  = dts[m * T + t];
      float c_t = gf * s_cd[q] + gi * gz;
      float cbt = gfb * s_cb[q] + gib * gz;
      float cdt = cbt + (c_t - cbt) * __expf(-gd * dt);
      float hdt = go * tanh_(cdt);
      s_cd[q] = cdt;
      s_cb[q] = cbt;
      int j = wg * NJ + jl;
      float* o = out + ((size_t)m * (T + 1) + (t + 1)) * OUT5;
      o[j]         = hdt;
      o[H + j]     = go;
      o[2 * H + j] = cbt;
      o[3 * H + j] = c_t;
      o[4 * H + j] = gd;
      hdn[m * H + j] = (bf16_t)hdt;
    }

    // Pre-compute next step's input-K partial sums while waiting on stragglers.
    if (t + 1 < T)
      inputGemm(rfrag + (((size_t)((t + 1) * 2 + mt) * KT_IN) << 9) + lane * 16,
                wbase, ip0, ip1, ip2, ip3);

    gridBarrier(bar, NWG);   // publishes h_d slice; includes __syncthreads
    pb ^= 1;
  }
}

// ---- launch ----------------------------------------------------------------
extern "C" void kernel_launch(void* const* d_in, const int* in_sizes, int n_in,
                              void* d_out, int out_size, void* d_ws, size_t ws_size,
                              hipStream_t stream) {
  const int*   marks  = (const int*)d_in[0];
  const float* ts     = (const float*)d_in[1];
  const float* latent = (const float*)d_in[2];
  const float* emb    = (const float*)d_in[3];
  const float* wcell  = (const float*)d_in[4];
  const float* bcell  = (const float*)d_in[5];
  const float* winit  = (const float*)d_in[6];
  const float* binit  = (const float*)d_in[7];
  float* out = (float*)d_out;

  char* ws = (char*)d_ws;
  unsigned* bar = (unsigned*)(ws + OFF_BAR);
  float*  dts = (float*)(ws + OFF_DT);
  bf16_t* hd  = (bf16_t*)(ws + OFF_HD);
  float*  cd0 = (float*)(ws + OFF_CD);
  float*  cb0 = (float*)(ws + OFF_CB);
  bf16_t* rf  = (bf16_t*)(ws + OFF_RF);
  bf16_t* wf  = (bf16_t*)(ws + OFF_WF);

  k_zero  <<<1, 64, 0, stream>>>(bar);
  k_dt    <<<(B * T + 255) / 256, 256, 0, stream>>>(ts, dts);
  k_pack_r<<<T * 2 * KT_IN, 512, 0, stream>>>(marks, latent, emb, rf);
  k_pack_w<<<NWG * 7 * KT, 512, 0, stream>>>(wcell, wf);
  k_init  <<<dim3(6 * H / 128, B), 128, 0, stream>>>(latent, winit, binit, out, hd, cd0, cb0);
  hawkes_recurrent<<<NWG, BLK, 0, stream>>>(rf, wf, dts, bcell, hd, cd0, cb0, out, bar);
}
